// TempEmbedder_3745211482418
// MI455X (gfx1250) — matrix operand order repaired and checked by reading, not compiled, not run on hardware
//
#include <hip/hip_runtime.h>
#include <hip/hip_bf16.h>
#include <stdint.h>

typedef __attribute__((ext_vector_type(16))) __bf16 v16bf;
typedef __attribute__((ext_vector_type(2)))  __bf16 bf16x2;
typedef __attribute__((ext_vector_type(8)))  float  v8f;
typedef unsigned short u16t;
typedef unsigned int   u32t;
typedef int v4i __attribute__((vector_size(16)));   // matches builtin's param type

union FragBF { v16bf v; uint4 q[2]; u32t u[8]; };
union AccF   { v8f  v; float f[8]; };

// ---- constants ----
#define BB 32
#define LL 2048
#define DIN 512
#define HH 256
#define OO 128
#define TOT ((size_t)BB * LL * OO)

// ---- CDNA5 async global->LDS path (ASYNCcnt, no VGPR round-trip) ----
#if defined(__AMDGCN__) && __has_builtin(__builtin_amdgcn_global_load_async_to_lds_b128)
#define HAVE_ASYNC_LDS 1
#else
#define HAVE_ASYNC_LDS 0
#endif

#define ASG(p) ((__attribute__((address_space(1))) v4i*)(uintptr_t)(p))
#define ASL(p) ((__attribute__((address_space(3))) v4i*)(uintptr_t)(p))

__device__ __forceinline__ void async_wait0() {
#if HAVE_ASYNC_LDS
#if __has_builtin(__builtin_amdgcn_s_wait_asynccnt)
    __builtin_amdgcn_s_wait_asynccnt(0);
#else
    asm volatile("s_wait_asynccnt 0x0" ::: "memory");
#endif
#endif
}

// hardware bf16 converts (v_cvt_*bf16_f32 on gfx1250) instead of bit-twiddling
__device__ __forceinline__ u16t f2bf(float f) {
    __bf16 h = (__bf16)f;
    return __builtin_bit_cast(u16t, h);
}
__device__ __forceinline__ u32t pk2(float lo, float hi) {
    bf16x2 t;
    t[0] = (__bf16)lo;
    t[1] = (__bf16)hi;
    return __builtin_bit_cast(u32t, t);
}
__device__ __forceinline__ float gelu_exact(float x) {
    return 0.5f * x * (1.0f + erff(x * 0.70710678118654752f));
}

// ---------------- weight conversion f32 -> bf16 (into workspace) ----------------
__global__ __launch_bounds__(256) void cvt_kernel(
    const float* __restrict__ W1, const float* __restrict__ W2,
    const float* __restrict__ W3, const float* __restrict__ W4,
    u16t* __restrict__ out)
{
    const int n1 = HH * DIN, n2 = HH * HH, n3 = OO * HH, n4 = OO * OO;
    int i = blockIdx.x * 256 + threadIdx.x;
    if (i < n1)                     out[i] = f2bf(W1[i]);
    else if (i < n1 + n2)           out[i] = f2bf(W2[i - n1]);
    else if (i < n1 + n2 + n3)      out[i] = f2bf(W3[i - n1 - n2]);
    else if (i < n1 + n2 + n3 + n4) out[i] = f2bf(W4[i - n1 - n2 - n3]);
}

// ---- A fragments from global f32 X (layer 1).  ISA 16-bit A 16x32 layout:
// lane<16: K = c*32 + {0..7, 16..23}; lane>=16: +8 on both runs. ----
__device__ __forceinline__ void load_a_global(FragBF* A, const float* __restrict__ xbase, int lane) {
    const int row = lane & 15;
    const int hi  = (lane >> 4) & 1;
    const float* xr = xbase + (size_t)row * DIN;
#pragma unroll
    for (int c = 0; c < DIN / 32; ++c) {
        int k0 = c * 32 + hi * 8;
        float4 p0 = *(const float4*)(xr + k0);
        float4 p1 = *(const float4*)(xr + k0 + 4);
        float4 p2 = *(const float4*)(xr + k0 + 16);
        float4 p3 = *(const float4*)(xr + k0 + 20);
        A[c].u[0] = pk2(p0.x, p0.y); A[c].u[1] = pk2(p0.z, p0.w);
        A[c].u[2] = pk2(p1.x, p1.y); A[c].u[3] = pk2(p1.z, p1.w);
        A[c].u[4] = pk2(p2.x, p2.y); A[c].u[5] = pk2(p2.z, p2.w);
        A[c].u[6] = pk2(p3.x, p3.y); A[c].u[7] = pk2(p3.z, p3.w);
    }
}

// ---- A fragments from per-wave LDS H buffer (row-major 16 x K bf16) ----
template <int K>
__device__ __forceinline__ void load_a_lds(FragBF* A, const u16t* h, int lane) {
    const int row = lane & 15;
    const int hi  = (lane >> 4) & 1;
#pragma unroll
    for (int c = 0; c < K / 32; ++c) {
        int off0 = row * K + c * 32 + hi * 8;       // ushort units; 16B aligned
        A[c].q[0] = *(const uint4*)(h + off0);
        A[c].q[1] = *(const uint4*)(h + off0 + 16);
    }
}

// ---- cooperative 16xK bf16 slab copy, 128 threads, 16B granularity ----
// Async path: tracked by ASYNCcnt, so the WMMA loop's s_wait_dscnt never drains it.
template <int K>
__device__ __forceinline__ void coop_copy(u16t* dst, const u16t* __restrict__ src, int tid) {
#if HAVE_ASYNC_LDS
#pragma unroll
    for (int t = 0; t < (16 * K) / 8 / 128; ++t) {
        int e = (tid + t * 128) * 8;                 // ushort index, 16B-aligned
        __builtin_amdgcn_global_load_async_to_lds_b128(ASG(src + e), ASL(dst + e), 0, 0);
    }
#else
    const uint4* s = (const uint4*)src;
    uint4* d = (uint4*)dst;
#pragma unroll
    for (int t = 0; t < (16 * K) / 8 / 128; ++t) d[tid + t * 128] = s[tid + t * 128];
#endif
}

// ---- one MLP layer: D(16xN) = A(16xK) * W^T tile-by-tile via WMMA bf16 ----
// B 32x16 layout (matches A): lane<16 holds col n, K=c*32+{0..7, 16..23};
// lane>=16 holds col n, +8 on both runs.
// Weight slab double-buffered; prefetch of tile j+1 overlaps tile j's WMMA stream.
template <int K, int N, bool GELU, bool LAST>
__device__ __forceinline__ void run_layer(
    const FragBF* A, const u16t* __restrict__ wb, const float* __restrict__ bias,
    u16t* wslab, u16t* hout, float* __restrict__ yout, int tid, int lane)
{
    const int n  = lane & 15;
    const int hi = (lane >> 4) & 1;
    coop_copy<K>(wslab, wb, tid);                       // prefetch tile 0 into buf0
    for (int j = 0; j < N / 16; ++j) {
        async_wait0();                                   // own async slab writes done
        __syncthreads();                                 // slab j visible to all waves
        const u16t* cur = wslab + (j & 1) * (16 * K);
        if (j + 1 < N / 16)                              // async prefetch next tile
            coop_copy<K>(wslab + ((j + 1) & 1) * (16 * K),
                         wb + (size_t)(j + 1) * 16 * K, tid);
        AccF acc;
        acc.v = (v8f){0.f, 0.f, 0.f, 0.f, 0.f, 0.f, 0.f, 0.f};
#pragma unroll
        for (int c = 0; c < K / 32; ++c) {
            FragBF b;
            int off0 = n * K + c * 32 + hi * 8;
            b.q[0] = *(const uint4*)(cur + off0);
            b.q[1] = *(const uint4*)(cur + off0 + 16);
            acc.v = __builtin_amdgcn_wmma_f32_16x16x32_bf16(
                false, A[c].v, false, b.v, (short)0, acc.v, false, false);
        }
        float bv = bias[j * 16 + n];
#pragma unroll
        for (int r = 0; r < 8; ++r) {
            int tokenRow = r + hi * 8;               // D layout: vgpr r -> M=r (+8 for hi half)
            float x = acc.f[r] + bv;
            if constexpr (GELU) x = gelu_exact(x);
            if constexpr (LAST) {
                yout[(size_t)tokenRow * OO + j * 16 + n] = x;
            } else {
                hout[tokenRow * N + j * 16 + n] = f2bf(x);
            }
        }
    }
}

// ---------------- fused 4-layer MLP: 4 waves, 16 tokens/wave ----------------
__global__ __launch_bounds__(128) void mlp_kernel(
    const float* __restrict__ X, const u16t* __restrict__ Wb,
    const float* __restrict__ b1, const float* __restrict__ b2,
    const float* __restrict__ b3, const float* __restrict__ b4,
    float* __restrict__ Y)
{
    __shared__ u16t wslab[2 * 16 * DIN];      // 32 KB double-buffered weight slab
    __shared__ u16t hbuf[4][16 * HH];         // 4 waves x 8 KB
    const int tid  = threadIdx.x;
    const int wave = tid >> 5;
    const int lane = tid & 31;
    const size_t tokenBase = (size_t)blockIdx.x * 64 + (size_t)wave * 16;

    const u16t* W1b = Wb;
    const u16t* W2b = W1b + HH * DIN;
    const u16t* W3b = W2b + HH * HH;
    const u16t* W4b = W3b + OO * HH;
    u16t* h = hbuf[wave];

    FragBF A[16];
    load_a_global(A, X + tokenBase * DIN, lane);
    run_layer<DIN, HH, true,  false>(A, W1b, b1, wslab, h, nullptr, tid, lane);
    load_a_lds<HH>(A, h, lane);
    run_layer<HH,  HH, true,  false>(A, W2b, b2, wslab, h, nullptr, tid, lane);
    load_a_lds<HH>(A, h, lane);
    run_layer<HH,  OO, true,  false>(A, W3b, b3, wslab, h, nullptr, tid, lane);
    load_a_lds<OO>(A, h, lane);
    run_layer<OO,  OO, false, true >(A, W4b, b4, wslab, nullptr, Y + tokenBase * OO, tid, lane);
}

// ---------------- sliding means over y (L2-resident), halo-tiled in LDS ----------------
// Halo is zero-filled, so all taps are unconditional; windows nest -> incremental sums
// (16 taps total). Slab fill uses the async global->LDS path where available.
__global__ __launch_bounds__(256) void slide_kernel(
    const float* __restrict__ Y, float* __restrict__ out /* = d_out + TOT (lw1 base) */)
{
    const int TL = 64, HALO = 8;
    __shared__ float slab[(TL + 2 * HALO) * OO];     // 40 KB
    const int b  = blockIdx.x >> 5;                  // 32 L-tiles per batch row
    const int lt = blockIdx.x & 31;
    const int l0 = lt * TL;
    const float* ybase = Y + (size_t)b * LL * OO;

    for (int t = threadIdx.x; t < (TL + 2 * HALO) * OO / 4; t += 256) {
        int idx = t * 4;
        int lr  = idx / OO;                 // 0..79
        int g   = l0 - HALO + lr;
        if (g >= 0 && g < LL) {
            const float* src = ybase + (size_t)g * OO + (idx & (OO - 1));
#if HAVE_ASYNC_LDS
            __builtin_amdgcn_global_load_async_to_lds_b128(ASG(src), ASL(slab + idx), 0, 0);
#else
            *(float4*)(slab + idx) = *(const float4*)src;
#endif
        } else {
            *(float4*)(slab + idx) = (float4){0.f, 0.f, 0.f, 0.f};
        }
    }
    async_wait0();
    __syncthreads();

    for (int p = 0; p < (TL * OO) / 256; ++p) {
        int idx = threadIdx.x + p * 256;
        int o   = idx & (OO - 1);
        int ll  = idx >> 7;                 // 0..63
        int l   = l0 + ll;
        const float* col = slab + (ll + HALO) * OO + o;   // tap d -> col[d*OO]
        size_t obase = ((size_t)b * LL + l) * OO + o;

        float s = col[-1 * OO] + col[0];                                    // [l-1, l+1)
        out[0 * TOT + obase] = s / (float)(min(l + 1, LL) - max(l - 1, 0));
        s += col[-2 * OO] + col[1 * OO];                                    // [l-2, l+2)
        out[1 * TOT + obase] = s / (float)(min(l + 2, LL) - max(l - 2, 0));
        s += col[-4 * OO] + col[-3 * OO] + col[2 * OO] + col[3 * OO];       // [l-4, l+4)
        out[2 * TOT + obase] = s / (float)(min(l + 4, LL) - max(l - 4, 0));
        s += col[-8 * OO] + col[-7 * OO] + col[-6 * OO] + col[-5 * OO]
           + col[ 4 * OO] + col[ 5 * OO] + col[ 6 * OO] + col[ 7 * OO];     // [l-8, l+8)
        out[3 * TOT + obase] = s / (float)(min(l + 8, LL) - max(l - 8, 0));
    }
}

extern "C" void kernel_launch(void* const* d_in, const int* in_sizes, int n_in,
                              void* d_out, int out_size, void* d_ws, size_t ws_size,
                              hipStream_t stream) {
    const float* X  = (const float*)d_in[0];
    const float* W1 = (const float*)d_in[1];
    const float* b1 = (const float*)d_in[2];
    const float* W2 = (const float*)d_in[3];
    const float* b2 = (const float*)d_in[4];
    const float* W3 = (const float*)d_in[5];
    const float* b3 = (const float*)d_in[6];
    const float* W4 = (const float*)d_in[7];
    const float* b4 = (const float*)d_in[8];
    float* out = (float*)d_out;
    u16t* wb = (u16t*)d_ws;   // 245,760 bf16 = 480 KB scratch

    cvt_kernel<<<(HH * DIN + HH * HH + OO * HH + OO * OO + 255) / 256, 256, 0, stream>>>(
        W1, W2, W3, W4, wb);
    mlp_kernel<<<(BB * LL) / 64, 128, 0, stream>>>(X, wb, b1, b2, b3, b4, out);
    slide_kernel<<<BB * (LL / 64), 256, 0, stream>>>(out, out + TOT);
}
